// WordTransformer_22101901705768
// MI455X (gfx1250) — matrix operand order repaired and checked by reading, not compiled, not run on hardware
//
#include <hip/hip_runtime.h>
#include <hip/hip_bf16.h>

// ---------------------------------------------------------------------------
// WordTransformer greedy decode for MI455X (gfx1250, wave32, WMMA).
//  * cross-attention collapsed to per-batch constant (memory is broadcast)
//  * all projections / FFN via v_wmma_f32_16x16x32_f16, f32 accumulate
//  * GEMM: A tile staged to LDS once per block via async global->LDS copy
//  * FFN fused through LDS, 32 tokens/block to amortize weight traffic
// ---------------------------------------------------------------------------

#define B_SZ    4096
#define D_MODEL 64
#define NHEADS  4
#define DH      16
#define NLAYERS 4
#define MAXLEN  12
#define VOCAB   29
#define DFF     2048
#define LNEPS   1e-5f
#define ATT_SCALE 0.25f   // 1/sqrt(16)

typedef __attribute__((ext_vector_type(16))) _Float16 v16h;
typedef __attribute__((ext_vector_type(8)))  _Float16 v8h;
typedef __attribute__((ext_vector_type(8)))  float    v8f;

// ----- async global->LDS (GLOBAL_LOAD_ASYNC_TO_LDS_B128, ASYNCcnt) ---------
// Probe result (round 2 diagnostic): builtin takes (v4i AS1*, v4i AS3*, imm, imm)
#if __has_builtin(__builtin_amdgcn_global_load_async_to_lds_b128)
#define HAVE_ASYNC_LDS 1
typedef int v4i_vs __attribute__((vector_size(16)));
__device__ __forceinline__ void async_copy_b128(const _Float16* g, _Float16* l) {
  __builtin_amdgcn_global_load_async_to_lds_b128(
      (__attribute__((address_space(1))) v4i_vs*)(g),
      (__attribute__((address_space(3))) v4i_vs*)(l), 0, 0);
}
__device__ __forceinline__ void wait_async0() {
#if __has_builtin(__builtin_amdgcn_s_wait_asynccnt)
  __builtin_amdgcn_s_wait_asynccnt(0);
#else
  asm volatile("s_wait_asynccnt 0x0" ::: "memory");
#endif
}
#else
#define HAVE_ASYNC_LDS 0
#endif

// ---------------- WMMA fragment helpers (wave32) ---------------------------
// A-matrix 16x32 f16 (ISA 7.12.2): lane<16 -> M=lane, K = {0..7,16..23};
// lane>=16 -> M=lane-16, K = {8..15,24..31}. Two 16B contiguous runs/lane.
__device__ __forceinline__ v16h load_frag_a(const _Float16* base, int row0,
                                            int rowMax, int kk, int lda) {
  const int lane = threadIdx.x & 31;
  int r = row0 + (lane & 15);
  if (r > rowMax) r = rowMax;
  const int kb = (lane & 16) ? 8 : 0;
  const _Float16* p = base + (size_t)r * lda + kk + kb;
  v8h lo = *(const v8h*)(p);
  v8h hi = *(const v8h*)(p + 16);
  v16h f;
#pragma unroll
  for (int j = 0; j < 8; ++j) { f[j] = lo[j]; f[8 + j] = hi[j]; }
  return f;
}

// B-matrix 32x16 f16 (matches A): lane<16 -> N=lane, K = {0..7,16..23};
// lane>=16 -> N=lane-16, K = {8..15,24..31}. Two 16B contiguous runs/lane.
// Weights stored [N,K] row-major so B[k][n] = W[n*ldb + k].
__device__ __forceinline__ v16h load_frag_b(const _Float16* base, int col0,
                                            int colMax, int kk, int ldb) {
  const int lane = threadIdx.x & 31;
  int c = col0 + (lane & 15);
  if (c > colMax) c = colMax;
  const int kb = (lane & 16) ? 8 : 0;
  const _Float16* p = base + (size_t)c * ldb + kk + kb;
  v8h lo = *(const v8h*)(p);
  v8h hi = *(const v8h*)(p + 16);
  v16h f;
#pragma unroll
  for (int j = 0; j < 8; ++j) { f[j] = lo[j]; f[8 + j] = hi[j]; }
  return f;
}

__device__ __forceinline__ v8f wmma_f16(v16h a, v16h b, v8f c) {
  return __builtin_amdgcn_wmma_f32_16x16x32_f16(false, a, false, b,
                                                (short)0, c, false, false);
}

// ---------------- generic GEMM: C = act(A[M,K] @ W[N,K]^T + bias) ----------
// block = 8 waves = one 16-row M tile x 8 N tiles. The shared 16xK A tile is
// staged into LDS once per block (async global->LDS when available), then all
// waves read A fragments from LDS (ds_load) and B fragments from global.
// Requires K <= 64, M multiple of 16.
__global__ __launch_bounds__(256) void gemm_wmma_kernel(
    const _Float16* __restrict__ A, int lda,
    const _Float16* __restrict__ W,
    const float* __restrict__ bias,
    float* __restrict__ C32, int ldc32,
    _Float16* __restrict__ C16, int ldc16,
    int M, int N, int K, int relu) {
  __shared__ _Float16 ldsA[16 * 64];
  const int wid  = threadIdx.x >> 5;
  const int lane = threadIdx.x & 31;
  const int tm   = blockIdx.y;
  const int tn   = blockIdx.x * 8 + wid;
  const int tilesN = (N + 15) >> 4;

  // stage A tile: 2K 16-byte chunks (K=64 -> 128 chunks, threads 0..127)
  {
    const int tid = threadIdx.x;
    const int nchunk = 2 * K;                 // 16 rows * K halves / 8
    if (tid < nchunk) {
      const int row = tid / (K >> 3);
      const int j   = tid % (K >> 3);
      const _Float16* gp = A + (size_t)(tm * 16 + row) * lda + j * 8;
      _Float16* lp = &ldsA[row * 64 + j * 8];
#if HAVE_ASYNC_LDS
      async_copy_b128(gp, lp);
#else
      *(v8h*)lp = *(const v8h*)gp;
#endif
    }
#if HAVE_ASYNC_LDS
    wait_async0();
#endif
    __syncthreads();
  }

  if (tn < tilesN) {                         // wave-uniform guard
    v8f acc = {};
    for (int kk = 0; kk < K; kk += 32) {
      v16h a = load_frag_a(ldsA, 0, 15, kk, 64);
      v16h b = load_frag_b(W, tn * 16, N - 1, kk, K);
      acc = wmma_f16(a, b, acc);
    }
    // C/D layout: lane<16 -> N=lane, M=tm*16+r ; lane>=16 -> N+.., M=+8+r
    const int col = tn * 16 + (lane & 15);
    const int rb  = tm * 16 + ((lane & 16) ? 8 : 0);
    const float bv = (bias && col < N) ? bias[col] : 0.f;
#pragma unroll
    for (int r = 0; r < 8; ++r) {
      float v = acc[r] + bv;
      if (relu) v = fmaxf(v, 0.f);
      if (col < N) {
        if (C32) C32[(size_t)(rb + r) * ldc32 + col] = v;
        if (C16) C16[(size_t)(rb + r) * ldc16 + col] = (_Float16)v;
      }
    }
  }
}

// ---------------- fused FFN: out = relu(h@W1^T+b1)@W2^T+b2 -----------------
// 4 waves / block, 32 tokens / block (2 M tiles -> weight traffic halved).
// relu(ff1) chunk [32x128] staged in LDS, consumed immediately into lin2.
__global__ __launch_bounds__(128) void ffn_fused_kernel(
    const _Float16* __restrict__ h16,
    const _Float16* __restrict__ W1,  // [2048,64]
    const float* __restrict__ b1,
    const _Float16* __restrict__ W2,  // [64,2048]
    const float* __restrict__ b2,
    float* __restrict__ out32,        // [M,64]
    int M) {
  __shared__ _Float16 lds_ff[32 * 128];
  const int wid  = threadIdx.x >> 5;   // 0..3
  const int lane = threadIdx.x & 31;
  const int row0 = blockIdx.x * 32;
  const int lrow0 = (lane & 16) ? 8 : 0;

  v16h a0[2], a1[2];
#pragma unroll
  for (int m = 0; m < 2; ++m) {
    a0[m] = load_frag_a(h16, row0 + m * 16, M - 1, 0, D_MODEL);
    a1[m] = load_frag_a(h16, row0 + m * 16, M - 1, 32, D_MODEL);
  }
  v8f acc2[2] = {{}, {}};              // lin2 tiles: output cols wid*16..+16

  for (int c = 0; c < DFF / 128; ++c) {
    if (c + 1 < DFF / 128) {           // prefetch next W1/W2 chunk rows
      __builtin_prefetch(W1 + (size_t)((c + 1) * 128 + wid * 32) * D_MODEL, 0, 1);
      __builtin_prefetch(W2 + (size_t)(wid * 16) * DFF + (c + 1) * 128, 0, 1);
    }
#pragma unroll
    for (int s = 0; s < 2; ++s) {      // each wave: 2 x 16-col ff1 tiles
      const int n0 = c * 128 + wid * 32 + s * 16;
      v16h b0  = load_frag_b(W1, n0, DFF - 1, 0, D_MODEL);
      v16h b1f = load_frag_b(W1, n0, DFF - 1, 32, D_MODEL);
      const int lcol = wid * 32 + s * 16 + (lane & 15);
      const float bias1 = b1[n0 + (lane & 15)];
#pragma unroll
      for (int m = 0; m < 2; ++m) {
        v8f acc = {};
        acc = wmma_f16(a0[m], b0, acc);
        acc = wmma_f16(a1[m], b1f, acc);
#pragma unroll
        for (int r = 0; r < 8; ++r) {
          float v = fmaxf(acc[r] + bias1, 0.f);
          lds_ff[(m * 16 + lrow0 + r) * 128 + lcol] = (_Float16)v;
        }
      }
    }
    __syncthreads();
#pragma unroll
    for (int kk = 0; kk < 128; kk += 32) {
      v16h bf = load_frag_b(W2, wid * 16, D_MODEL - 1, c * 128 + kk, DFF);
#pragma unroll
      for (int m = 0; m < 2; ++m) {
        v16h af = load_frag_a(lds_ff + m * 16 * 128, 0, 15, kk, 128);
        acc2[m] = wmma_f16(af, bf, acc2[m]);
      }
    }
    __syncthreads();
  }
  const int col = wid * 16 + (lane & 15);
  const float bias2 = b2[col];
#pragma unroll
  for (int m = 0; m < 2; ++m) {
    const int rb = row0 + m * 16 + lrow0;
#pragma unroll
    for (int r = 0; r < 8; ++r)
      out32[(size_t)(rb + r) * D_MODEL + col] = acc2[m][r] + bias2;
  }
}

// ---------------- small kernels --------------------------------------------
__global__ void f32_to_f16_kernel(const float* __restrict__ s,
                                  _Float16* __restrict__ d, int n) {
  int i = blockIdx.x * blockDim.x + threadIdx.x;
  if (i < n) d[i] = (_Float16)s[i];
}

__global__ void inembed_kernel(const float* __restrict__ x,
                               const float* __restrict__ w1,
                               const float* __restrict__ b1,
                               _Float16* __restrict__ hid16) {
  int i = blockIdx.x * blockDim.x + threadIdx.x;   // B*64
  if (i >= B_SZ * D_MODEL) return;
  int d = i & 63, b = i >> 6;
  hid16[i] = (_Float16)fmaxf(x[b] * w1[d] + b1[d], 0.f);
}

__global__ void dec_init_kernel(int* __restrict__ dec) {
  int b = blockIdx.x * blockDim.x + threadIdx.x;
  if (b < B_SZ) dec[b * (MAXLEN + 1)] = 1;          // start token id 1
}

__global__ void embed_kernel(const int* __restrict__ dec,
                             const float* __restrict__ tok_emb,
                             const float* __restrict__ pos_enc,
                             float* __restrict__ h32,
                             _Float16* __restrict__ h16, int T) {
  int i = blockIdx.x * blockDim.x + threadIdx.x;   // B*T*64
  if (i >= B_SZ * T * D_MODEL) return;
  int d = i & 63, tp = i >> 6, s = tp % T, b = tp / T;
  int tok = dec[b * (MAXLEN + 1) + s];
  float v = tok_emb[tok * D_MODEL + d] + pos_enc[s * D_MODEL + d];
  h32[i] = v; h16[i] = (_Float16)v;
}

// self-attention (no causal mask, per the source): one thread per (b,q,head)
__global__ void attention_kernel(const _Float16* __restrict__ qkv, // [B*T,192]
                                 _Float16* __restrict__ attnout,   // [B*T,64]
                                 int T) {
  int i = blockIdx.x * blockDim.x + threadIdx.x;   // B*T*NHEADS
  if (i >= B_SZ * T * NHEADS) return;
  int hd = i & 3, q = (i >> 2) % T, b = i / (T * NHEADS);
  const _Float16* base = qkv + (size_t)(b * T) * 192;
  const _Float16* qp = base + (size_t)q * 192 + hd * DH;
  float qv[DH];
#pragma unroll
  for (int j = 0; j < DH; ++j) qv[j] = (float)qp[j];
  float sc[MAXLEN], mx = -1e30f;
  for (int k = 0; k < T; ++k) {
    const _Float16* kp = base + (size_t)k * 192 + 64 + hd * DH;
    float s = 0.f;
#pragma unroll
    for (int j = 0; j < DH; ++j) s += qv[j] * (float)kp[j];
    s *= ATT_SCALE; sc[k] = s; mx = fmaxf(mx, s);
  }
  float o[DH]; float den = 0.f;
#pragma unroll
  for (int j = 0; j < DH; ++j) o[j] = 0.f;
  for (int k = 0; k < T; ++k) {
    float w = __expf(sc[k] - mx); den += w;
    const _Float16* vp = base + (size_t)k * 192 + 128 + hd * DH;
#pragma unroll
    for (int j = 0; j < DH; ++j) o[j] += w * (float)vp[j];
  }
  float inv = 1.f / den;
  _Float16* op = attnout + (size_t)(b * T + q) * D_MODEL + hd * DH;
#pragma unroll
  for (int j = 0; j < DH; ++j) op[j] = (_Float16)(o[j] * inv);
}

// h = LN(h + add); one token per wave (wave32: 2 elems/lane, shfl reduction)
__global__ void residual_ln_kernel(float* __restrict__ h32,
                                   _Float16* __restrict__ h16,
                                   const float* __restrict__ add,
                                   int addPerBatch, int T,
                                   const float* __restrict__ g,
                                   const float* __restrict__ bta, int ntok) {
  int wid = threadIdx.x >> 5, lane = threadIdx.x & 31;
  int tok = blockIdx.x * (blockDim.x >> 5) + wid;
  if (tok >= ntok) return;
  size_t base  = (size_t)tok * D_MODEL;
  size_t abase = addPerBatch ? (size_t)(tok / T) * D_MODEL : base;
  float v0 = h32[base + lane]      + add[abase + lane];
  float v1 = h32[base + 32 + lane] + add[abase + 32 + lane];
  float s = v0 + v1;
#pragma unroll
  for (int off = 16; off; off >>= 1) s += __shfl_xor(s, off, 32);
  float mean = s * (1.f / 64.f);
  float d0 = v0 - mean, d1 = v1 - mean;
  float q = d0 * d0 + d1 * d1;
#pragma unroll
  for (int off = 16; off; off >>= 1) q += __shfl_xor(q, off, 32);
  float inv = rsqrtf(q * (1.f / 64.f) + LNEPS);
  float y0 = d0 * inv * g[lane]      + bta[lane];
  float y1 = d1 * inv * g[lane + 32] + bta[lane + 32];
  h32[base + lane] = y0;      h32[base + 32 + lane] = y1;
  h16[base + lane] = (_Float16)y0; h16[base + 32 + lane] = (_Float16)y1;
}

__global__ void argmax_kernel(const float* __restrict__ logits, int ldl,
                              int* __restrict__ dec, int t) {
  int b = blockIdx.x * blockDim.x + threadIdx.x;
  if (b >= B_SZ) return;
  const float* p = logits + (size_t)b * ldl;
  float best = p[0]; int bi = 0;
  for (int v = 1; v < VOCAB; ++v) { float f = p[v]; if (f > best) { best = f; bi = v; } }
  dec[b * (MAXLEN + 1) + t + 1] = bi;
}

// ---------------------------------------------------------------------------
static void launch_gemm(hipStream_t st, const _Float16* A, int lda,
                        const _Float16* W, const float* bias,
                        float* C32, int ldc32, _Float16* C16, int ldc16,
                        int M, int N, int K, int relu) {
  dim3 grid(((N + 15) / 16 + 7) / 8, M / 16);
  gemm_wmma_kernel<<<grid, 256, 0, st>>>(A, lda, W, bias, C32, ldc32,
                                         C16, ldc16, M, N, K, relu);
}

extern "C" void kernel_launch(void* const* d_in, const int* in_sizes, int n_in,
                              void* d_out, int out_size, void* d_ws, size_t ws_size,
                              hipStream_t stream) {
  (void)in_sizes; (void)n_in; (void)out_size; (void)ws_size;
  const float* x        = (const float*)d_in[0];
  const float* tok_emb  = (const float*)d_in[1];
  const float* pos_enc  = (const float*)d_in[2];
  const float* in1_w    = (const float*)d_in[3];
  const float* in1_b    = (const float*)d_in[4];
  const float* in2_w    = (const float*)d_in[5];
  const float* in2_b    = (const float*)d_in[6];
  const float* sa_qkv_w = (const float*)d_in[7];
  const float* sa_qkv_b = (const float*)d_in[8];
  const float* sa_out_w = (const float*)d_in[9];
  const float* sa_out_b = (const float*)d_in[10];
  const float* ca_qkv_w = (const float*)d_in[11];
  const float* ca_qkv_b = (const float*)d_in[12];
  const float* ca_out_w = (const float*)d_in[13];
  const float* ca_out_b = (const float*)d_in[14];
  const float* lin1_w   = (const float*)d_in[15];
  const float* lin1_b   = (const float*)d_in[16];
  const float* lin2_w   = (const float*)d_in[17];
  const float* lin2_b   = (const float*)d_in[18];
  const float* ln1_g    = (const float*)d_in[19];
  const float* ln1_b    = (const float*)d_in[20];
  const float* ln2_g    = (const float*)d_in[21];
  const float* ln2_b    = (const float*)d_in[22];
  const float* ln3_g    = (const float*)d_in[23];
  const float* ln3_b    = (const float*)d_in[24];
  const float* out_w    = (const float*)d_in[25];
  const float* out_b    = (const float*)d_in[26];
  float* out = (float*)d_out;                       // [B, 12, 29]

  // -------- workspace bump allocator (256B aligned) --------
  char* wsb = (char*)d_ws;
  size_t off = 0;
  auto alloc = [&](size_t bytes) -> char* {
    char* p = wsb + off;
    off = (off + bytes + 255) & ~(size_t)255;
    return p;
  };
  _Float16* w16_in2   = (_Float16*)alloc(64 * 64 * 2);
  _Float16* w16_saqkv = (_Float16*)alloc((size_t)NLAYERS * 192 * 64 * 2);
  _Float16* w16_saout = (_Float16*)alloc((size_t)NLAYERS * 64 * 64 * 2);
  _Float16* w16_caqkv = (_Float16*)alloc((size_t)NLAYERS * 192 * 64 * 2);
  _Float16* w16_caout = (_Float16*)alloc((size_t)NLAYERS * 64 * 64 * 2);
  _Float16* w16_lin1  = (_Float16*)alloc((size_t)NLAYERS * DFF * 64 * 2);
  _Float16* w16_lin2  = (_Float16*)alloc((size_t)NLAYERS * 64 * DFF * 2);
  _Float16* w16_out   = (_Float16*)alloc((size_t)VOCAB * 64 * 2);
  _Float16* hid16     = (_Float16*)alloc((size_t)B_SZ * 64 * 2);
  _Float16* mem16     = (_Float16*)alloc((size_t)B_SZ * 64 * 2);
  _Float16* vmem16    = (_Float16*)alloc((size_t)B_SZ * 64 * 2);
  float*    ca_add32  = (float*)   alloc((size_t)NLAYERS * B_SZ * 64 * 4);
  int*      dec       = (int*)     alloc((size_t)B_SZ * (MAXLEN + 1) * 4);
  float*    h32       = (float*)   alloc((size_t)B_SZ * MAXLEN * 64 * 4);
  _Float16* h16       = (_Float16*)alloc((size_t)B_SZ * MAXLEN * 64 * 2);
  _Float16* qkv16     = (_Float16*)alloc((size_t)B_SZ * MAXLEN * 192 * 2);
  _Float16* attnout16 = (_Float16*)alloc((size_t)B_SZ * MAXLEN * 64 * 2);
  float*    sub32     = (float*)   alloc((size_t)B_SZ * MAXLEN * 64 * 4);

  auto cvt = [&](const float* s, _Float16* d, int n) {
    f32_to_f16_kernel<<<(n + 255) / 256, 256, 0, stream>>>(s, d, n);
  };
  // -------- weight conversion (once per launch) --------
  cvt(in2_w,    w16_in2,   64 * 64);
  cvt(sa_qkv_w, w16_saqkv, NLAYERS * 192 * 64);
  cvt(sa_out_w, w16_saout, NLAYERS * 64 * 64);
  cvt(ca_qkv_w, w16_caqkv, NLAYERS * 192 * 64);
  cvt(ca_out_w, w16_caout, NLAYERS * 64 * 64);
  cvt(lin1_w,   w16_lin1,  NLAYERS * DFF * 64);
  cvt(lin2_w,   w16_lin2,  NLAYERS * 64 * DFF);
  cvt(out_w,    w16_out,   VOCAB * 64);

  // -------- memory vector + collapsed cross-attention constants --------
  inembed_kernel<<<(B_SZ * 64 + 255) / 256, 256, 0, stream>>>(x, in1_w, in1_b, hid16);
  launch_gemm(stream, hid16, 64, w16_in2, in2_b,
              nullptr, 0, mem16, 64, B_SZ, 64, 64, 0);
  for (int l = 0; l < NLAYERS; ++l) {
    // v = mem @ Wv^T + bv   (Wv = rows 128..191 of packed ca in_proj)
    launch_gemm(stream, mem16, 64,
                w16_caqkv + (size_t)l * 192 * 64 + 128 * 64,
                ca_qkv_b + l * 192 + 128,
                nullptr, 0, vmem16, 64, B_SZ, 64, 64, 0);
    // ca_add[l] = v @ Wo^T + bo  (softmax over identical keys == identity)
    launch_gemm(stream, vmem16, 64, w16_caout + (size_t)l * 64 * 64,
                ca_out_b + l * 64,
                ca_add32 + (size_t)l * B_SZ * 64, 64, nullptr, 0,
                B_SZ, 64, 64, 0);
  }

  dec_init_kernel<<<(B_SZ + 255) / 256, 256, 0, stream>>>(dec);

  // -------- greedy decode --------
  for (int t = 0; t < MAXLEN; ++t) {
    const int T = t + 1;
    const int ntok = B_SZ * T;
    embed_kernel<<<(ntok * 64 + 255) / 256, 256, 0, stream>>>(
        dec, tok_emb, pos_enc, h32, h16, T);
    for (int l = 0; l < NLAYERS; ++l) {
      // self-attention qkv projection  [ntok,64] x [192,64]^T
      launch_gemm(stream, h16, 64, w16_saqkv + (size_t)l * 192 * 64,
                  sa_qkv_b + l * 192, nullptr, 0, qkv16, 192,
                  ntok, 192, 64, 0);
      attention_kernel<<<(ntok * NHEADS + 255) / 256, 256, 0, stream>>>(
          qkv16, attnout16, T);
      // out projection
      launch_gemm(stream, attnout16, 64, w16_saout + (size_t)l * 64 * 64,
                  sa_out_b + l * 64, sub32, 64, nullptr, 0,
                  ntok, 64, 64, 0);
      residual_ln_kernel<<<(ntok + 3) / 4, 128, 0, stream>>>(
          h32, h16, sub32, 0, T, ln1_g + l * 64, ln1_b + l * 64, ntok);
      // collapsed cross-attention: per-batch constant add
      residual_ln_kernel<<<(ntok + 3) / 4, 128, 0, stream>>>(
          h32, h16, ca_add32 + (size_t)l * B_SZ * 64, 1, T,
          ln2_g + l * 64, ln2_b + l * 64, ntok);
      // fused FFN through LDS (32 tokens per block)
      ffn_fused_kernel<<<ntok / 32, 128, 0, stream>>>(
          h16, w16_lin1 + (size_t)l * DFF * 64, lin1_b + l * DFF,
          w16_lin2 + (size_t)l * 64 * DFF, lin2_b + l * 64, sub32, ntok);
      residual_ln_kernel<<<(ntok + 3) / 4, 128, 0, stream>>>(
          h32, h16, sub32, 0, T, ln3_g + l * 64, ln3_b + l * 64, ntok);
    }
    // vocab head on last token: A rows stride T*64, base offset (T-1)*64
    launch_gemm(stream, h16 + (size_t)(T - 1) * 64, T * 64, w16_out, out_b,
                out + (size_t)t * VOCAB, MAXLEN * VOCAB, nullptr, 0,
                B_SZ, VOCAB, 64, 0);
    argmax_kernel<<<(B_SZ + 255) / 256, 256, 0, stream>>>(
        out + (size_t)t * VOCAB, MAXLEN * VOCAB, dec, t);
  }
}
